// RBF_Layer_67336497267088
// MI455X (gfx1250) — matrix operand order repaired and checked by reading, not compiled, not run on hardware
//
#include <hip/hip_runtime.h>
#include <hip/hip_bf16.h>

#define B_ROWS 65536
#define K_IN   128
#define N_OUT  512

#define NCHUNK 128   // centers columns staged in LDS per chunk (4 chunks, double-buffered)
#define PITCH  136   // padded bf16 pitch: 272B rows -> bank-conflict-free, 16B aligned

#define FB_NCHUNK 256  // fallback (no-workspace) kernel chunk

typedef __bf16 v16bf __attribute__((ext_vector_type(16)));
typedef __bf16 v8bf  __attribute__((ext_vector_type(8)));
typedef __bf16 v4bf  __attribute__((ext_vector_type(4)));
typedef float  v8f   __attribute__((ext_vector_type(8)));
typedef float  v4f   __attribute__((ext_vector_type(4)));

// ---------------- prep: c_sq[o] = sum_k centers[o,k]^2 ----------------
__global__ __launch_bounds__(256)
void rbf_csq_kernel(const float* __restrict__ centers, float* __restrict__ csq) {
    int o = blockIdx.x * blockDim.x + threadIdx.x;
    if (o < N_OUT) {
        const v4f* row = (const v4f*)(centers + o * K_IN);
        float s = 0.f;
        #pragma unroll
        for (int i = 0; i < K_IN / 4; ++i) {
            v4f v = row[i];
            s += v[0]*v[0] + v[1]*v[1] + v[2]*v[2] + v[3]*v[3];
        }
        csq[o] = s;
    }
}

// ---------------- prep: centers fp32 -> bf16 hi/lo split (once, into ws) ----------------
__global__ __launch_bounds__(256)
void rbf_cvt_kernel(const float* __restrict__ centers,
                    __bf16* __restrict__ cHi, __bf16* __restrict__ cLo) {
    int g = blockIdx.x * blockDim.x + threadIdx.x;   // 16384 threads, 4 elems each
    v4f v = *(const v4f*)(centers + (size_t)g * 4);
    v4bf hi4, lo4;
    #pragma unroll
    for (int j = 0; j < 4; ++j) {
        __bf16 hv = (__bf16)v[j];
        hi4[j] = hv;
        lo4[j] = (__bf16)(v[j] - (float)hv);
    }
    *(v4bf*)(cHi + (size_t)g * 4) = hi4;
    *(v4bf*)(cLo + (size_t)g * 4) = lo4;
}

// ---------------- main (async-staged, double-buffered) ----------------
__global__ __launch_bounds__(256)
void rbf_main_async(const float* __restrict__ x,
                    const __bf16* __restrict__ cHi,
                    const __bf16* __restrict__ cLo,
                    const float* __restrict__ beta,
                    const float* __restrict__ csq,
                    float* __restrict__ out) {
    __shared__ __bf16 sB[2][2][NCHUNK * PITCH];   // [buffer][hi/lo][padded tile]

    const int tid   = threadIdx.x;
    const int lane  = tid & 31;
    const int wave  = tid >> 5;
    const int h     = lane >> 4;
    const int ln16  = lane & 15;
    const int mBase = blockIdx.x * 128 + wave * 16;
    const int row   = mBase + ln16;

    // async stage of one 128-col chunk of bf16 centers into padded LDS
    // (generic ptr low 32 bits == LDS offset, per aperture truncation rule)
    auto stage = [&](int chunk) {
        const char* srcH = (const char*)(cHi + (size_t)chunk * NCHUNK * K_IN);
        const char* srcL = (const char*)(cLo + (size_t)chunk * NCHUNK * K_IN);
        unsigned dH = (unsigned)(uintptr_t)(void*)&sB[chunk & 1][0][0];
        unsigned dL = (unsigned)(uintptr_t)(void*)&sB[chunk & 1][1][0];
        #pragma unroll
        for (int i = 0; i < (NCHUNK * 16) / 256; ++i) {  // 16 x b128 per 256B row
            int linear = i * 256 + tid;
            int col = linear >> 4;
            int j   = linear & 15;
            unsigned lofs = (unsigned)(col * (PITCH * 2) + j * 16);
            const char* gH = srcH + col * 256 + j * 16;
            const char* gL = srcL + col * 256 + j * 16;
            asm volatile("global_load_async_to_lds_b128 %0, %1, off"
                         :: "v"(dH + lofs), "v"(gH) : "memory");
            asm volatile("global_load_async_to_lds_b128 %0, %1, off"
                         :: "v"(dL + lofs), "v"(gL) : "memory");
        }
    };

    stage(0);   // overlap first fill with A-fragment setup below

    // ---- A fragments: 16 rows x 128 K per wave, fp32 -> bf16 hi/lo split ----
    v16bf aHi[4], aLo[4];
    float xsqPart = 0.f;
    #pragma unroll
    for (int c = 0; c < 4; ++c) {
        const float* base = x + (size_t)row * K_IN + c * 32 + h * 16;
        v4f r0 = *(const v4f*)(base);
        v4f r1 = *(const v4f*)(base + 4);
        v4f r2 = *(const v4f*)(base + 8);
        v4f r3 = *(const v4f*)(base + 12);
        float f[16];
        #pragma unroll
        for (int j = 0; j < 4; ++j) { f[j]=r0[j]; f[4+j]=r1[j]; f[8+j]=r2[j]; f[12+j]=r3[j]; }
        #pragma unroll
        for (int e = 0; e < 16; ++e) {
            float v = f[e];
            xsqPart += v * v;
            __bf16 hv = (__bf16)v;
            aHi[c][e] = hv;
            aLo[c][e] = (__bf16)(v - (float)hv);
        }
    }
    float xsq = xsqPart + __shfl_xor(xsqPart, 16, 32);
    float xsq8[8];
    #pragma unroll
    for (int r = 0; r < 8; ++r)
        xsq8[r] = __shfl(xsq, r + 8 * h, 32);

    const int NC = N_OUT / NCHUNK;   // 4 chunks
    for (int c = 0; c < NC; ++c) {
        asm volatile("s_wait_asynccnt 0x0" ::: "memory");  // my stage(c) done
        __syncthreads();                                   // everyone's stage(c) + compute(c-1) done
        if (c + 1 < NC) stage(c + 1);                      // prefetch into other buffer

        const __bf16* bhB = &sB[c & 1][0][0];
        const __bf16* blB = &sB[c & 1][1][0];
        #pragma unroll 1
        for (int t = 0; t < NCHUNK / 16; ++t) {
            int colL = t * 16 + ln16;
            int col  = c * NCHUNK + colL;
            v8f acc0 = {0.f,0.f,0.f,0.f,0.f,0.f,0.f,0.f};  // hi*hi chain
            v8f acc1 = {0.f,0.f,0.f,0.f,0.f,0.f,0.f,0.f};  // hi*lo + lo*hi chain
            #pragma unroll
            for (int cc = 0; cc < 4; ++cc) {
                const __bf16* bh = bhB + colL * PITCH + cc * 32 + h * 16;
                const __bf16* bl = blB + colL * PITCH + cc * 32 + h * 16;
                v8bf bh0 = *(const v8bf*)bh;
                v8bf bh1 = *(const v8bf*)(bh + 8);
                v8bf bl0 = *(const v8bf*)bl;
                v8bf bl1 = *(const v8bf*)(bl + 8);
                v16bf bHi = __builtin_shufflevector(bh0, bh1, 0,1,2,3,4,5,6,7,8,9,10,11,12,13,14,15);
                v16bf bLo = __builtin_shufflevector(bl0, bl1, 0,1,2,3,4,5,6,7,8,9,10,11,12,13,14,15);
                acc0 = __builtin_amdgcn_wmma_f32_16x16x32_bf16(false, aHi[cc], false, bHi,
                                                               (short)0, acc0, false, false);
                acc1 = __builtin_amdgcn_wmma_f32_16x16x32_bf16(false, aHi[cc], false, bLo,
                                                               (short)0, acc1, false, false);
                acc1 = __builtin_amdgcn_wmma_f32_16x16x32_bf16(false, aLo[cc], false, bHi,
                                                               (short)0, acc1, false, false);
            }
            float cs = csq[col];
            float bt = beta[col];
            float* op = out + (size_t)(mBase + 8 * h) * N_OUT + col;
            #pragma unroll
            for (int r = 0; r < 8; ++r) {
                float d2 = fmaxf(xsq8[r] + cs - 2.0f * (acc0[r] + acc1[r]), 0.0f);
                op[(size_t)r * N_OUT] = __expf(-bt * d2);
            }
        }
    }
}

// ---------------- fallback main (no workspace for converted centers) ----------------
__global__ __launch_bounds__(256)
void rbf_main_fallback(const float* __restrict__ x,
                       const float* __restrict__ centers,
                       const float* __restrict__ beta,
                       const float* __restrict__ csq,
                       float* __restrict__ out) {
    __shared__ __bf16 sHi[FB_NCHUNK * PITCH];
    __shared__ __bf16 sLo[FB_NCHUNK * PITCH];

    const int tid   = threadIdx.x;
    const int lane  = tid & 31;
    const int wave  = tid >> 5;
    const int h     = lane >> 4;
    const int ln16  = lane & 15;
    const int mBase = blockIdx.x * 128 + wave * 16;
    const int row   = mBase + ln16;

    v16bf aHi[4], aLo[4];
    float xsqPart = 0.f;
    #pragma unroll
    for (int c = 0; c < 4; ++c) {
        const float* base = x + (size_t)row * K_IN + c * 32 + h * 16;
        v4f r0 = *(const v4f*)(base);
        v4f r1 = *(const v4f*)(base + 4);
        v4f r2 = *(const v4f*)(base + 8);
        v4f r3 = *(const v4f*)(base + 12);
        float f[16];
        #pragma unroll
        for (int j = 0; j < 4; ++j) { f[j]=r0[j]; f[4+j]=r1[j]; f[8+j]=r2[j]; f[12+j]=r3[j]; }
        #pragma unroll
        for (int e = 0; e < 16; ++e) {
            float v = f[e];
            xsqPart += v * v;
            __bf16 hv = (__bf16)v;
            aHi[c][e] = hv;
            aLo[c][e] = (__bf16)(v - (float)hv);
        }
    }
    float xsq = xsqPart + __shfl_xor(xsqPart, 16, 32);
    float xsq8[8];
    #pragma unroll
    for (int r = 0; r < 8; ++r)
        xsq8[r] = __shfl(xsq, r + 8 * h, 32);

    for (int nc = 0; nc < N_OUT; nc += FB_NCHUNK) {
        #pragma unroll 1
        for (int i = tid; i < FB_NCHUNK * K_IN / 4; i += 256) {
            int lin = i * 4;
            int col = lin / K_IN;
            int k   = lin % K_IN;
            v4f v = *(const v4f*)(centers + (size_t)(nc + col) * K_IN + k);
            v4bf hi4, lo4;
            #pragma unroll
            for (int j = 0; j < 4; ++j) {
                __bf16 hv = (__bf16)v[j];
                hi4[j] = hv;
                lo4[j] = (__bf16)(v[j] - (float)hv);
            }
            *(v4bf*)(sHi + col * PITCH + k) = hi4;
            *(v4bf*)(sLo + col * PITCH + k) = lo4;
        }
        __syncthreads();

        #pragma unroll 1
        for (int t = 0; t < FB_NCHUNK / 16; ++t) {
            int colL = t * 16 + ln16;
            int col  = nc + colL;
            v8f acc0 = {0.f,0.f,0.f,0.f,0.f,0.f,0.f,0.f};
            v8f acc1 = {0.f,0.f,0.f,0.f,0.f,0.f,0.f,0.f};
            #pragma unroll
            for (int cc = 0; cc < 4; ++cc) {
                const __bf16* bh = sHi + colL * PITCH + cc * 32 + h * 16;
                const __bf16* bl = sLo + colL * PITCH + cc * 32 + h * 16;
                v8bf bh0 = *(const v8bf*)bh;
                v8bf bh1 = *(const v8bf*)(bh + 8);
                v8bf bl0 = *(const v8bf*)bl;
                v8bf bl1 = *(const v8bf*)(bl + 8);
                v16bf bHi = __builtin_shufflevector(bh0, bh1, 0,1,2,3,4,5,6,7,8,9,10,11,12,13,14,15);
                v16bf bLo = __builtin_shufflevector(bl0, bl1, 0,1,2,3,4,5,6,7,8,9,10,11,12,13,14,15);
                acc0 = __builtin_amdgcn_wmma_f32_16x16x32_bf16(false, aHi[cc], false, bHi,
                                                               (short)0, acc0, false, false);
                acc1 = __builtin_amdgcn_wmma_f32_16x16x32_bf16(false, aHi[cc], false, bLo,
                                                               (short)0, acc1, false, false);
                acc1 = __builtin_amdgcn_wmma_f32_16x16x32_bf16(false, aLo[cc], false, bHi,
                                                               (short)0, acc1, false, false);
            }
            float cs = csq[col];
            float bt = beta[col];
            float* op = out + (size_t)(mBase + 8 * h) * N_OUT + col;
            #pragma unroll
            for (int r = 0; r < 8; ++r) {
                float d2 = fmaxf(xsq8[r] + cs - 2.0f * (acc0[r] + acc1[r]), 0.0f);
                op[(size_t)r * N_OUT] = __expf(-bt * d2);
            }
        }
        __syncthreads();
    }
}

extern "C" void kernel_launch(void* const* d_in, const int* in_sizes, int n_in,
                              void* d_out, int out_size, void* d_ws, size_t ws_size,
                              hipStream_t stream) {
    (void)in_sizes; (void)n_in; (void)out_size;
    const float* x       = (const float*)d_in[0];
    const float* centers = (const float*)d_in[1];
    const float* beta    = (const float*)d_in[2];
    float* out = (float*)d_out;

    const size_t needBf = (size_t)N_OUT * K_IN * sizeof(__bf16);        // per hi/lo array
    const size_t need   = 2 * needBf + N_OUT * sizeof(float);

    if (ws_size >= need) {
        __bf16* cHi = (__bf16*)d_ws;
        __bf16* cLo = cHi + (size_t)N_OUT * K_IN;
        float*  csq = (float*)(cLo + (size_t)N_OUT * K_IN);
        rbf_cvt_kernel<<<(N_OUT * K_IN / 4 + 255) / 256, 256, 0, stream>>>(centers, cHi, cLo);
        rbf_csq_kernel<<<(N_OUT + 255) / 256, 256, 0, stream>>>(centers, csq);
        rbf_main_async<<<B_ROWS / 128, 256, 0, stream>>>(x, cHi, cLo, beta, csq, out);
    } else {
        float* csq = (float*)d_ws;   // needs only 2 KB
        rbf_csq_kernel<<<(N_OUT + 255) / 256, 256, 0, stream>>>(centers, csq);
        rbf_main_fallback<<<B_ROWS / 128, 256, 0, stream>>>(x, centers, beta, csq, out);
    }
}
